// ShiftWiseConv_13176959664664
// MI455X (gfx1250) — matrix operand order repaired and checked by reading, not compiled, not run on hardware
//
#include <hip/hip_runtime.h>

typedef __bf16  v16bf __attribute__((ext_vector_type(16)));
typedef float   v8f   __attribute__((ext_vector_type(8)));
typedef int     i32x4 __attribute__((vector_size(16)));   // matches builtin param type

#define BM 128
#define BN 96
#define BK 32
#define C1 192
#define C2 192
#define KK 13
#define HW 56
#define HP 68            // 56 + 2*6 padded
#define NB 8
#define MTOT (NB*HW*HW)  // 25088
#define NKSTEPS (KK*KK*(C1/BK))  // 169*6 = 1014
#define EPSV 1e-5f

// Uniform K-walk deltas (elements)
#define D_CB   (BK)                                    // within a (kh,kw) position
#define D_KW   (BK)                                    // C1 - (C1-BK)
#define D_KH   (HP*C1 - (KK-1)*C1 - (C1-BK))           // 10592
#define D_BPOS (C2*C1 - (C1-BK))                       // 36704

#if __has_builtin(__builtin_amdgcn_global_load_async_to_lds_b128)
#define HAVE_ASYNC_LDS 1
#else
#define HAVE_ASYNC_LDS 0
#endif

struct U4x2 { uint4 a, b; };

static __device__ inline v16bf frag_cast(uint4 a, uint4 b) {
  U4x2 t; t.a = a; t.b = b;
  return __builtin_bit_cast(v16bf, t);
}

static __device__ inline unsigned short f2bf(float v) {
  unsigned u = __builtin_bit_cast(unsigned, v);
  unsigned r = (u + 0x7FFFu + ((u >> 16) & 1u)) >> 16;  // round-to-nearest-even
  return (unsigned short)r;
}

// 16-byte async Global->LDS copy (CDNA5 async engine, ASYNCcnt-tracked).
static __device__ inline void async_cp16(const void* g, void* l) {
#if HAVE_ASYNC_LDS
  __builtin_amdgcn_global_load_async_to_lds_b128(
      (__attribute__((address_space(1))) i32x4*)(unsigned long long)g,
      (__attribute__((address_space(3))) i32x4*)(unsigned int)(unsigned long long)l,
      0, 0);
#else
  *(uint4*)l = *(const uint4*)g;
#endif
}

static __device__ inline void async_fence() {
#if HAVE_ASYNC_LDS
#if __has_builtin(__builtin_amdgcn_s_wait_asynccnt)
  __builtin_amdgcn_s_wait_asynccnt(0);
#else
  asm volatile("s_wait_asynccnt 0" ::: "memory");
#endif
#endif
}

// fp32 NCHW -> bf16 padded NHWC  (8 x 68 x 68 x 192), zero borders
__global__ __launch_bounds__(256) void pack_x(const float* __restrict__ x,
                                              unsigned short* __restrict__ xp) {
  int i = blockIdx.x * 256 + threadIdx.x;          // 8*68*68*192 = 7,102,464 total
  int c = i % C1; int t = i / C1;
  int xx = t % HP; t /= HP;
  int yy = t % HP; int b = t / HP;
  int ih = yy - 6, iw = xx - 6;
  float v = 0.0f;
  if (ih >= 0 && ih < HW && iw >= 0 && iw < HW)
    v = x[(((size_t)b * C1 + c) * HW + ih) * HW + iw];
  xp[i] = f2bf(v);
}

// fp32 OIHW -> bf16 [kpos][co][ci]
__global__ __launch_bounds__(256) void pack_w(const float* __restrict__ w,
                                              unsigned short* __restrict__ wq) {
  int i = blockIdx.x * 256 + threadIdx.x;          // 169*192*192 = 6,230,016 total
  int ci = i % C1; int t = i / C1;
  int co = t % C2; int kpos = t / C2;
  wq[i] = f2bf(w[((size_t)co * C1 + ci) * (KK * KK) + kpos]);
}

__global__ __launch_bounds__(256) void conv_bn_silu_gemm(
    const unsigned short* __restrict__ xp,
    const unsigned short* __restrict__ wq,
    const float* __restrict__ gamma,
    const float* __restrict__ beta,
    const float* __restrict__ rmean,
    const float* __restrict__ rvar,
    float* __restrict__ out)
{
  __shared__ uint4 Ab[2][BM * 4];   // 128 rows x 64B
  __shared__ uint4 Bb[2][BN * 4];   //  96 rows x 64B

  const int tid   = threadIdx.x;
  const int bid   = blockIdx.x;
  const int ntile = bid & 1;
  const int mtile = bid >> 1;
  const int lane  = tid & 31;
  const int wv    = tid >> 5;
  const int wm    = wv & 3;    // wave m position (0..3), 2 m-frags each
  const int wn    = wv >> 2;   // wave n position (0..1), 3 n-frags each
  const int half  = lane >> 4;
  const int l15   = lane & 15;

  v8f acc[2][3] = {};

  // A-row geometry is loop-invariant per thread
  const int ar    = tid >> 1;
  const int apart = tid & 1;
  const int am    = mtile * BM + ar;
  const int abatch = am / (HW * HW);
  const int as    = am - abatch * (HW * HW);
  const int aoh   = as / HW;
  const int aow   = as - aoh * HW;

  // Loop-invariant per-thread base pointers; all advancing state is UNIFORM
  // (SGPR-resident) 32-bit offsets -> nothing per-thread to spill.
  const unsigned short* const aBase = xp +
      ((size_t)(abatch * HP + aoh) * HP + aow) * C1 + apart * 16;
  const unsigned short* const bBase = wq +
      (size_t)(ntile * BN + (tid >> 1)) * C1 + apart * 16;   // valid for tid < 192
  const unsigned short* const pBase = aBase;

  // LDS destinations are loop-invariant too (double-buffer index flips).
  uint4* const aDst = &Ab[0][ar * 4 + apart * 2];
  uint4* const bDst = &Bb[0][(tid >> 1) * 4 + apart * 2];

  unsigned koff = 0, boff = 0;     // uniform element offsets for stage stream
  int cb = 0, kw = 0;              // uniform K-walk counters
  unsigned poff = 2 * D_CB;        // prefetch stream runs 2 steps ahead (ks=2)
  int pcb = 2 * BK, pkw = 0;

  // ---- prologue: stage ks = 0 into buffer 0 ----
  async_cp16(aBase + koff,     aDst + 0);
  async_cp16(aBase + koff + 8, aDst + 1);
  if (tid < 2 * BN) {
    async_cp16(bBase + boff,     bDst + 0);
    async_cp16(bBase + boff + 8, bDst + 1);
  }
  // advance to ks = 1
  cb += BK; koff += D_CB; boff += D_CB;

  async_fence();
  __syncthreads();

  for (int ks = 0; ks < NKSTEPS; ++ks) {
    const int cur = ks & 1;
    const unsigned bufw = (cur ^ 1) * (BM * 4);   // word offset of the write buffer (A)
    const unsigned bufb = (cur ^ 1) * (BN * 4);

    // Fragment gathers per ISA 16-bit layouts:
    // A: lane<16 holds K0-7 (bytes 0-15) + K16-23 (bytes 32-47); lane>=16 the other halves.
    v16bf af[2], bf[3];
    #pragma unroll
    for (int i = 0; i < 2; ++i) {
      const int row = (wm * 2 + i) * 16 + l15;
      af[i] = frag_cast(Ab[cur][row * 4 + half], Ab[cur][row * 4 + 2 + half]);
    }
    // B: same K split as A: lane<16 holds K0-7 + K16-23; lane>=16 holds K8-15 + K24-31.
    #pragma unroll
    for (int j = 0; j < 3; ++j) {
      const int nr = (wn * 3 + j) * 16 + l15;
      bf[j] = frag_cast(Bb[cur][nr * 4 + half], Bb[cur][nr * 4 + 2 + half]);
    }

    if (ks + 1 < NKSTEPS) {       // async DMA overlaps the WMMAs below
      async_cp16(aBase + koff,     aDst + bufw + 0);
      async_cp16(aBase + koff + 8, aDst + bufw + 1);
      if (tid < 2 * BN) {
        async_cp16(bBase + boff,     bDst + bufb + 0);
        async_cp16(bBase + boff + 8, bDst + bufb + 1);
      }
      // uniform advance (SALU)
      cb += BK;
      if (cb != C1) {
        koff += D_CB; boff += D_CB;
      } else {
        cb = 0; boff += D_BPOS;
        if (++kw != KK) koff += D_KW;
        else { kw = 0; koff += D_KH; }
      }
    }

    if (ks + 2 < NKSTEPS) {       // software prefetch -> global_prefetch_b8
      __builtin_prefetch(pBase + poff, 0, 3);
      pcb += BK;
      if (pcb != C1) poff += D_CB;
      else {
        pcb = 0;
        if (++pkw != KK) poff += D_KW;
        else { pkw = 0; poff += D_KH; }
      }
    }

    #pragma unroll
    for (int i = 0; i < 2; ++i)
      #pragma unroll
      for (int j = 0; j < 3; ++j)
        acc[i][j] = __builtin_amdgcn_wmma_f32_16x16x32_bf16(
            false, af[i], false, bf[j], (short)0, acc[i][j], false, false);

    async_fence();      // next buffer's DMA must land before the barrier
    __syncthreads();
  }

  // Fused BatchNorm (inference affine) + SiLU epilogue, write NCHW fp32.
  #pragma unroll
  for (int j = 0; j < 3; ++j) {
    const int co = ntile * BN + (wn * 3 + j) * 16 + l15;
    const float sc = gamma[co] * rsqrtf(rvar[co] + EPSV);
    const float sh = beta[co] - sc * rmean[co];
    #pragma unroll
    for (int i = 0; i < 2; ++i) {
      const int mb = mtile * BM + (wm * 2 + i) * 16 + half * 8;
      #pragma unroll
      for (int r = 0; r < 8; ++r) {
        const int m = mb + r;
        const int b = m / (HW * HW);
        const int s = m - b * (HW * HW);
        const float y = acc[i][j][r] * sc + sh;
        out[((size_t)b * C2 + co) * (HW * HW) + s] = y / (1.0f + __expf(-y));
      }
    }
  }
}

extern "C" void kernel_launch(void* const* d_in, const int* in_sizes, int n_in,
                              void* d_out, int out_size, void* d_ws, size_t ws_size,
                              hipStream_t stream) {
  const float* x     = (const float*)d_in[0];
  const float* w     = (const float*)d_in[1];
  const float* gamma = (const float*)d_in[2];
  const float* beta  = (const float*)d_in[3];
  const float* rmean = (const float*)d_in[4];
  const float* rvar  = (const float*)d_in[5];
  float* out = (float*)d_out;

  unsigned short* xp = (unsigned short*)d_ws;                       // 14.2 MB
  unsigned short* wq = xp + (size_t)NB * HP * HP * C1;              // 12.5 MB

  pack_x<<<dim3((NB * HP * HP * C1) / 256), dim3(256), 0, stream>>>(x, xp);
  pack_w<<<dim3((KK * KK * C1 * C2) / 256), dim3(256), 0, stream>>>(w, wq);
  conv_bn_silu_gemm<<<dim3((MTOT / BM) * 2), dim3(256), 0, stream>>>(
      xp, wq, gamma, beta, rmean, rvar, out);
}
